// Greedy_GLM_38319698215447
// MI455X (gfx1250) — matrix operand order repaired and checked by reading, not compiled, not run on hardware
//
#include <hip/hip_runtime.h>
#include <hip/hip_bf16.h>

typedef __attribute__((ext_vector_type(16))) _Float16 v16h;
typedef __attribute__((ext_vector_type(8)))  float    v8f;

#define SUB_NO 20
#define T_NO   201
#define E_NO   2000
#define I_NO   500
#define T_DATA 20000
#define EPAD   2016   // E_NO rounded up to multiple of 32
#define IPAD   512    // I_NO rounded up to multiple of 32

// ---------------------------------------------------------------------------
// Kernel A: column softmax over 20 rows; writes f32 output + padded f16 copy
// (32 rows x Kpad cols, zeros in padding) for use as WMMA B operand.
// ---------------------------------------------------------------------------
__global__ void softmax_pack_kernel(const float* __restrict__ raw, int Kc, int Kpad,
                                    float* __restrict__ outF32,
                                    _Float16* __restrict__ outF16)
{
    int k = blockIdx.x * blockDim.x + threadIdx.x;
    if (k >= Kpad) return;
    if (k >= Kc) {
        for (int n = 0; n < 32; ++n) outF16[n * Kpad + k] = (_Float16)0.f;
        return;
    }
    float mx = raw[k];
    for (int n = 1; n < SUB_NO; ++n) mx = fmaxf(mx, raw[n * Kc + k]);
    float ex[SUB_NO];
    float s = 0.f;
    for (int n = 0; n < SUB_NO; ++n) {
        ex[n] = expf(raw[n * Kc + k] - mx);
        s += ex[n];
    }
    float inv = 1.f / s;
    for (int n = 0; n < SUB_NO; ++n) {
        float v = ex[n] * inv;
        outF32[n * Kc + k]  = v;
        outF16[n * Kpad + k] = (_Float16)v;
    }
    for (int n = SUB_NO; n < 32; ++n) outF16[n * Kpad + k] = (_Float16)0.f;
}

// ---------------------------------------------------------------------------
// Kernel B: synaptic filter kernels e_kern / i_kern -> d_out filter block
// out layout: rows 0..19 = e_kern[i][tau], rows 20..39 = i_kern[i][tau]
// ---------------------------------------------------------------------------
__global__ void filter_kernel(const float* __restrict__ W_syn,
                              const float* __restrict__ Tau_syn,
                              const float* __restrict__ Delta_syn,
                              float* __restrict__ outF)
{
    int tid = blockIdx.x * blockDim.x + threadIdx.x;
    if (tid >= SUB_NO * T_NO) return;
    int i = tid / T_NO;
    int tau = tid - i * T_NO;
    float de = expf(Delta_syn[i * 2 + 0]);
    float di = expf(Delta_syn[i * 2 + 1]);
    float te = expf(Tau_syn[i * 2 + 0]);
    float ti = expf(Tau_syn[i * 2 + 1]);
    float we = expf(W_syn[i * 2 + 0]);
    float wi = expf(W_syn[i * 2 + 1]);
    float t_e = fmaxf((float)tau - de, 0.f);
    float t_i = fmaxf((float)tau - di, 0.f);
    float tte = t_e / te;
    float tti = t_i / ti;
    outF[i * T_NO + tau]            =  tte * expf(-tte) * we;
    outF[(SUB_NO + i) * T_NO + tau] = -(tti * expf(-tti) * wi);
}

// ---------------------------------------------------------------------------
// Kernel C: syn = S (MxK f32) @ Cpad^T via v_wmma_f32_16x16x32_f16.
// One wave per 16-row M tile; N=20 covered by two 16-wide WMMA accumulators
// (tile 1 columns 20..31 are zero-padded in Cpad). f32->f16 conversion of S
// happens in-flight so S is streamed from HBM exactly once.
// ---------------------------------------------------------------------------
__global__ __launch_bounds__(256)
void wmma_syn_gemm(const float* __restrict__ S, const _Float16* __restrict__ Cpad,
                   int K, int Kpad, float* __restrict__ out, int Mtiles)
{
    int wave  = threadIdx.x >> 5;
    int lane  = threadIdx.x & 31;
    int mtile = blockIdx.x * (blockDim.x >> 5) + wave;
    if (mtile >= Mtiles) return;

    int half = lane >> 4;    // 0: lanes 0-15, 1: lanes 16-31
    int lrow = lane & 15;

    const float*    Srow  = S + (size_t)(mtile * 16 + lrow) * (size_t)K;
    const _Float16* B0row = Cpad + (size_t)lrow        * (size_t)Kpad + half * 16;
    const _Float16* B1row = Cpad + (size_t)(16 + lrow) * (size_t)Kpad + half * 16;

    v8f c0 = {};
    v8f c1 = {};

    int kfull = (K >> 5) << 5;
    for (int k0 = 0; k0 < kfull; k0 += 32) {
        // A 16x32 f16 k order used here (same for B): lanes 0-15 hold K k0+[0..15];
        // lanes 16-31 hold K k0+16+[0..15].
        const float4* p0 = (const float4*)(Srow + k0 + half * 16);
        const float4* p1 = (const float4*)(Srow + k0 + half * 16 + 8);
        float4 f0 = p0[0], f1 = p0[1];
        float4 f2 = p1[0], f3 = p1[1];
        v16h a;
        a[0]  = (_Float16)f0.x; a[1]  = (_Float16)f0.y; a[2]  = (_Float16)f0.z; a[3]  = (_Float16)f0.w;
        a[4]  = (_Float16)f1.x; a[5]  = (_Float16)f1.y; a[6]  = (_Float16)f1.z; a[7]  = (_Float16)f1.w;
        a[8]  = (_Float16)f2.x; a[9]  = (_Float16)f2.y; a[10] = (_Float16)f2.z; a[11] = (_Float16)f2.w;
        a[12] = (_Float16)f3.x; a[13] = (_Float16)f3.y; a[14] = (_Float16)f3.z; a[15] = (_Float16)f3.w;

        v16h b0 = *(const v16h*)(B0row + k0);
        v16h b1 = *(const v16h*)(B1row + k0);

        c0 = __builtin_amdgcn_wmma_f32_16x16x32_f16(false, a, false, b0, (short)0, c0, false, false);
        c1 = __builtin_amdgcn_wmma_f32_16x16x32_f16(false, a, false, b1, (short)0, c1, false, false);
    }

    if (kfull < K) {   // K remainder (16 for E_NO, 20 for I_NO): guarded scalar A
        v16h a;
        #pragma unroll
        for (int j = 0; j < 16; ++j) {
            int k = kfull + half * 16 + j;
            a[j] = (k < K) ? (_Float16)Srow[k] : (_Float16)0.f;
        }
        v16h b0 = *(const v16h*)(B0row + kfull);   // Cpad zero-padded to Kpad
        v16h b1 = *(const v16h*)(B1row + kfull);
        c0 = __builtin_amdgcn_wmma_f32_16x16x32_f16(false, a, false, b0, (short)0, c0, false, false);
        c1 = __builtin_amdgcn_wmma_f32_16x16x32_f16(false, a, false, b1, (short)0, c1, false, false);
    }

    // C/D layout: lane L, vgpr r -> M = r + (L>=16 ? 8 : 0), N = L&15
    float* orow = out + (size_t)(mtile * 16 + half * 8) * SUB_NO;
    #pragma unroll
    for (int r = 0; r < 8; ++r) {
        orow[r * SUB_NO + lrow] = c0[r];
        if (lrow < SUB_NO - 16) orow[r * SUB_NO + 16 + lrow] = c1[r];
    }
}

// ---------------------------------------------------------------------------
// Kernel D: causal 201-tap filter + dendritic tanh tree + output V.
// Per block: stage a 328x20 halo tile of syn_e/syn_i plus both filter banks
// into LDS (~89 KB of the WGP's 320 KB), then each thread owns one timestep.
// ---------------------------------------------------------------------------
#define CT_TB 128
#define CT_ROWS (CT_TB + T_NO - 1)   // 328
#define CT_STR 21                    // row stride (coprime with 64 banks)

__global__ __launch_bounds__(CT_TB)
void conv_tree_kernel(const float* __restrict__ syn_e, const float* __restrict__ syn_i,
                      const float* __restrict__ filt,  const float* __restrict__ W_sub,
                      const float* __restrict__ Theta, const int* __restrict__ Cden,
                      const float* __restrict__ V_o,   float* __restrict__ V)
{
    __shared__ float se[CT_ROWS * CT_STR];
    __shared__ float si[CT_ROWS * CT_STR];
    __shared__ float fe[SUB_NO * T_NO];
    __shared__ float fi[SUB_NO * T_NO];
    __shared__ float gain[SUB_NO];
    __shared__ float th[SUB_NO];
    __shared__ int   cd[SUB_NO * SUB_NO];

    int tid = threadIdx.x;
    int t0  = blockIdx.x * CT_TB;

    if (tid < SUB_NO) {
        gain[tid] = expf(W_sub[tid]);
        th[tid]   = Theta[tid];
    }
    for (int idx = tid; idx < SUB_NO * SUB_NO; idx += CT_TB) cd[idx] = Cden[idx];
    for (int idx = tid; idx < SUB_NO * T_NO; idx += CT_TB) {
        fe[idx] = filt[idx];
        fi[idx] = filt[SUB_NO * T_NO + idx];
    }
    for (int idx = tid; idx < CT_ROWS * SUB_NO; idx += CT_TB) {
        int rr = idx / SUB_NO;
        int i  = idx - rr * SUB_NO;
        int g  = t0 - (T_NO - 1) + rr;
        float ve = 0.f, vi = 0.f;
        if (g >= 0 && g < T_DATA) {
            ve = syn_e[g * SUB_NO + i];
            vi = syn_i[g * SUB_NO + i];
        }
        se[rr * CT_STR + i] = ve;
        si[rr * CT_STR + i] = vi;
    }
    __syncthreads();

    int t = t0 + tid;
    if (t >= T_DATA) return;

    float syn[SUB_NO];
    #pragma unroll
    for (int i = 0; i < SUB_NO; ++i) syn[i] = 0.f;

    for (int tau = 0; tau < T_NO; ++tau) {
        int r = (tid + (T_NO - 1) - tau) * CT_STR;
        #pragma unroll
        for (int i = 0; i < SUB_NO; ++i) {
            syn[i] += fe[i * T_NO + tau] * se[r + i] + fi[i * T_NO + tau] * si[r + i];
        }
    }

    float sub[SUB_NO];
    #pragma unroll
    for (int i = 0; i < SUB_NO; ++i) sub[i] = 0.f;
    for (int i = SUB_NO - 1; i >= 0; --i) {
        float child = 0.f;
        for (int j = 0; j < SUB_NO; ++j)
            if (cd[i * SUB_NO + j]) child += gain[j] * sub[j];
        sub[i] = tanhf(syn[i] + child + th[i]);
    }
    V[t] = gain[0] * sub[0] + V_o[0];
}

// ---------------------------------------------------------------------------
// Launch: softmax(E), softmax(I), filters, WMMA GEMM(E), WMMA GEMM(I), conv+tree
// d_out: V[20000] | filters[40*201] | C_syn_e[20*2000] | C_syn_i[20*500]
// d_ws : CpadE f16[32*2016] | CpadI f16[32*512] | syn_e f32[20000*20] | syn_i
// ---------------------------------------------------------------------------
extern "C" void kernel_launch(void* const* d_in, const int* in_sizes, int n_in,
                              void* d_out, int out_size, void* d_ws, size_t ws_size,
                              hipStream_t stream)
{
    const float* S_e       = (const float*)d_in[0];
    const float* S_i       = (const float*)d_in[1];
    const int*   C_den     = (const int*)  d_in[2];
    const float* W_syn     = (const float*)d_in[3];
    const float* Tau_syn   = (const float*)d_in[4];
    const float* Delta_syn = (const float*)d_in[5];
    const float* W_sub     = (const float*)d_in[6];
    const float* V_o       = (const float*)d_in[7];
    const float* Theta     = (const float*)d_in[8];
    const float* Ce_raw    = (const float*)d_in[9];
    const float* Ci_raw    = (const float*)d_in[10];

    float* out    = (float*)d_out;
    float* V      = out;
    float* filt   = out + T_DATA;
    float* Ce_out = out + T_DATA + 2 * SUB_NO * T_NO;
    float* Ci_out = Ce_out + SUB_NO * E_NO;

    char* ws = (char*)d_ws;
    _Float16* CpadE = (_Float16*)ws;                              // 129024 B
    _Float16* CpadI = (_Float16*)(ws + 32 * EPAD * 2);            //  32768 B
    float* syn_e    = (float*)(ws + 32 * EPAD * 2 + 32 * IPAD * 2);
    float* syn_i    = syn_e + (size_t)T_DATA * SUB_NO;            // total ~3.4 MB

    softmax_pack_kernel<<<(EPAD + 127) / 128, 128, 0, stream>>>(Ce_raw, E_NO, EPAD, Ce_out, CpadE);
    softmax_pack_kernel<<<(IPAD + 127) / 128, 128, 0, stream>>>(Ci_raw, I_NO, IPAD, Ci_out, CpadI);
    filter_kernel<<<(SUB_NO * T_NO + 127) / 128, 128, 0, stream>>>(W_syn, Tau_syn, Delta_syn, filt);

    int Mtiles = T_DATA / 16;   // 1250, exact
    wmma_syn_gemm<<<(Mtiles + 7) / 8, 256, 0, stream>>>(S_e, CpadE, E_NO, EPAD, syn_e, Mtiles);
    wmma_syn_gemm<<<(Mtiles + 7) / 8, 256, 0, stream>>>(S_i, CpadI, I_NO, IPAD, syn_i, Mtiles);

    conv_tree_kernel<<<(T_DATA + CT_TB - 1) / CT_TB, CT_TB, 0, stream>>>(
        syn_e, syn_i, filt, W_sub, Theta, C_den, V_o, V);
}
